// LinearModel_21311627723192
// MI455X (gfx1250) — 8-bit matrix operand order repaired and checked by reading, compiled, not run on hardware
//
#include <hip/hip_runtime.h>
#include <math.h>

typedef int v8i __attribute__((ext_vector_type(8)));

#define K1      784
#define K1V4    196
#define K1PAD   832
#define A_STR   836
#define ACT_STR 132

__device__ __forceinline__ float q8(float v) {
    return fminf(fmaxf(rintf(v), -128.0f), 127.0f);
}

__global__ __launch_bounds__(256) void compute_scales_kernel(
    const float* __restrict__ W1, const float* __restrict__ W2,
    const float* __restrict__ W3, float* __restrict__ scales) {
    __shared__ float red[256];
    const float* W; int n;
    if (blockIdx.x == 0)      { W = W1; n = 128 * 784; }
    else if (blockIdx.x == 1) { W = W2; n = 128 * 128; }
    else                      { W = W3; n = 10 * 128;  }
    float s = 0.0f;
    for (int i = threadIdx.x; i < n; i += 256) s += fabsf(W[i]);
    red[threadIdx.x] = s;
    __syncthreads();
    for (int off = 128; off > 0; off >>= 1) {
        if ((int)threadIdx.x < off) red[threadIdx.x] += red[threadIdx.x + off];
        __syncthreads();
    }
    if (threadIdx.x == 0) scales[blockIdx.x] = red[0] / (float)n;
}

__global__ __launch_bounds__(256) void pack_signs_kernel(
    const float* __restrict__ W1, const float* __restrict__ W2,
    const float* __restrict__ W3,
    signed char* __restrict__ sw1, signed char* __restrict__ sw2,
    signed char* __restrict__ sw3) {
    const int NW1 = 128 * K1PAD;
    const int NW2 = 128 * 128;
    const int NW3 = 16 * 128;
    int idx = blockIdx.x * 256 + threadIdx.x;
    if (idx < NW1) {
        int n = idx / K1PAD, k = idx - n * K1PAD;
        sw1[idx] = (k < K1) ? ((W1[n * K1 + k] >= 0.0f) ? 1 : -1) : 0;
    } else if (idx < NW1 + NW2) {
        int j = idx - NW1;
        sw2[j] = (W2[j] >= 0.0f) ? 1 : -1;
    } else if (idx < NW1 + NW2 + NW3) {
        int j = idx - NW1 - NW2;
        int n = j >> 7;
        sw3[j] = (n < 10) ? ((W3[j] >= 0.0f) ? 1 : -1) : 0;
    }
}

__global__ __launch_bounds__(256) void mlp_fused_kernel(
    const float* __restrict__ x,
    const signed char* __restrict__ sw1,
    const signed char* __restrict__ sw2,
    const signed char* __restrict__ sw3,
    const float* __restrict__ b1, const float* __restrict__ b2,
    const float* __restrict__ b3,
    const float* __restrict__ scales,
    const float* __restrict__ s_in,
    const float* __restrict__ s1_in, const float* __restrict__ s1_out,
    const float* __restrict__ s2_in, const float* __restrict__ s2_out,
    const float* __restrict__ s3_in, const float* __restrict__ s3_out,
    float* __restrict__ out) {

    __shared__ signed char aq[16 * A_STR];
    __shared__ signed char act1[16 * ACT_STR];
    __shared__ signed char act2[16 * ACT_STR];

    const int tid  = threadIdx.x;
    const int wave = tid >> 5;
    const int lane = tid & 31;
    const int m    = lane & 15;
    const int half = lane >> 4;
    const long row0 = (long)blockIdx.x * 16;

    const float inv_sin = 1.0f / s_in[0];

    for (int i = tid; i < 16 * K1V4; i += 256) {
        const int r  = i / K1V4;
        const int c4 = i - r * K1V4;
        const float4 v = ((const float4*)(x + (row0 + r) * K1))[c4];
        const int p0 = (int)q8(v.x * inv_sin) & 0xFF;
        const int p1 = (int)q8(v.y * inv_sin) & 0xFF;
        const int p2 = (int)q8(v.z * inv_sin) & 0xFF;
        const int p3 = (int)q8(v.w * inv_sin) & 0xFF;
        *(int*)(aq + r * A_STR + c4 * 4) = p0 | (p1 << 8) | (p2 << 16) | (p3 << 24);
    }
    if (tid < 16 * 12) {
        const int r = tid / 12, c4 = tid - r * 12;
        *(int*)(aq + r * A_STR + K1 + c4 * 4) = 0;
    }
    __syncthreads();

    {
        const signed char* bw = sw1 + (wave * 16 + m) * K1PAD;
        v8i acc = {0, 0, 0, 0, 0, 0, 0, 0};
        v8i bcur;
#pragma unroll
        for (int r = 0; r < 8; ++r) {
            const int colB = (r < 4) ? (half * 16 + r * 4)
                                     : (32 + half * 16 + (r - 4) * 4);
            bcur[r] = *(const int*)(bw + colB);
        }
        for (int kc = 0; kc < 13; ++kc) {
            v8i a;
#pragma unroll
            for (int r = 0; r < 8; ++r) {
                const int colA = kc * 64 + ((r < 4) ? (half * 16 + r * 4) : (32 + half * 16 + (r - 4) * 4));
                a[r] = *(const int*)(aq + m * A_STR + colA);
            }
            v8i bnext = bcur;
            if (kc < 12) {
#pragma unroll
                for (int r = 0; r < 8; ++r) {
                    const int colB = (kc + 1) * 64 +
                        ((r < 4) ? (half * 16 + r * 4)
                                 : (32 + half * 16 + (r - 4) * 4));
                    bnext[r] = *(const int*)(bw + colB);
                }
            }
            acc = __builtin_amdgcn_wmma_i32_16x16x64_iu8(true, a, true, bcur, acc,
                                                         false, false);
            bcur = bnext;
        }
        const float sc = s_in[0] * scales[0];
        const float qi = s1_in[0], qo = s1_out[0];
        const float inv_qi = 1.0f / qi, inv_qo = 1.0f / qo;
#pragma unroll
        for (int r = 0; r < 8; ++r) {
            const int mm = r + half * 8;
            const int nn = wave * 16 + m;
            float y  = (float)acc[r] * sc + b1[nn];
            float qa = q8(y * inv_qi) * qi;
            float rl = fmaxf(qa, 0.0f);
            act1[mm * ACT_STR + nn] = (signed char)(int)q8(rl * inv_qo);
        }
    }
    __syncthreads();

    {
        v8i acc = {0, 0, 0, 0, 0, 0, 0, 0};
        for (int kc = 0; kc < 2; ++kc) {
            v8i a, b;
#pragma unroll
            for (int r = 0; r < 8; ++r) {
                const int colA = kc * 64 + ((r < 4) ? (half * 16 + r * 4) : (32 + half * 16 + (r - 4) * 4));
                a[r] = *(const int*)(act1 + m * ACT_STR + colA);
                const int colB = kc * 64 + ((r < 4) ? (half * 16 + r * 4)
                                                    : (32 + half * 16 + (r - 4) * 4));
                b[r] = *(const int*)(sw2 + (wave * 16 + m) * 128 + colB);
            }
            acc = __builtin_amdgcn_wmma_i32_16x16x64_iu8(true, a, true, b, acc,
                                                         false, false);
        }
        const float sc = s1_out[0] * scales[1];
        const float qi = s2_in[0], qo = s2_out[0];
        const float inv_qi = 1.0f / qi, inv_qo = 1.0f / qo;
#pragma unroll
        for (int r = 0; r < 8; ++r) {
            const int mm = r + half * 8;
            const int nn = wave * 16 + m;
            float y  = (float)acc[r] * sc + b2[nn];
            float qa = q8(y * inv_qi) * qi;
            float rl = fmaxf(qa, 0.0f);
            act2[mm * ACT_STR + nn] = (signed char)(int)q8(rl * inv_qo);
        }
    }
    __syncthreads();

    if (wave == 0) {
        v8i acc = {0, 0, 0, 0, 0, 0, 0, 0};
        for (int kc = 0; kc < 2; ++kc) {
            v8i a, b;
#pragma unroll
            for (int r = 0; r < 8; ++r) {
                const int colA = kc * 64 + ((r < 4) ? (half * 16 + r * 4) : (32 + half * 16 + (r - 4) * 4));
                a[r] = *(const int*)(act2 + m * ACT_STR + colA);
                const int colB = kc * 64 + ((r < 4) ? (half * 16 + r * 4)
                                                    : (32 + half * 16 + (r - 4) * 4));
                b[r] = *(const int*)(sw3 + m * 128 + colB);
            }
            acc = __builtin_amdgcn_wmma_i32_16x16x64_iu8(true, a, true, b, acc,
                                                         false, false);
        }
        const float sc = s2_out[0] * scales[2];
        const float qi = s3_in[0], qo = s3_out[0];
        const float inv_qi = 1.0f / qi, inv_qo = 1.0f / qo;
        const float bias = (m < 10) ? b3[m] : 0.0f;
#pragma unroll
        for (int r = 0; r < 8; ++r) {
            const int mm = r + half * 8;
            float y  = (float)acc[r] * sc + bias;
            float z  = q8(y * inv_qi) * qi;
            float sg = 1.0f / (1.0f + expf(-z));
            float o  = q8(sg * inv_qo) * qo;
            if (m < 10) out[(row0 + mm) * 10 + m] = o;
        }
    }
}

extern "C" void kernel_launch(void* const* d_in, const int* in_sizes, int n_in,
                              void* d_out, int out_size, void* d_ws, size_t ws_size,
                              hipStream_t stream) {
    const float* x     = (const float*)d_in[0];
    const float* W1    = (const float*)d_in[1];
    const float* b1    = (const float*)d_in[2];
    const float* W2    = (const float*)d_in[3];
    const float* b2    = (const float*)d_in[4];
    const float* W3    = (const float*)d_in[5];
    const float* b3    = (const float*)d_in[6];
    const float* s_in  = (const float*)d_in[7];
    const float* s1_in = (const float*)d_in[8];
    const float* s1_out= (const float*)d_in[9];
    const float* s2_in = (const float*)d_in[10];
    const float* s2_out= (const float*)d_in[11];
    const float* s3_in = (const float*)d_in[12];
    const float* s3_out= (const float*)d_in[13];
    float* out = (float*)d_out;

    char* ws = (char*)d_ws;
    float*       scales = (float*)ws;
    signed char* sw1    = (signed char*)(ws + 16);
    signed char* sw2    = sw1 + 128 * K1PAD;
    signed char* sw3    = sw2 + 128 * 128;

    const int rows = in_sizes[0] / K1;
    const int packN = 128 * K1PAD + 128 * 128 + 16 * 128;

    compute_scales_kernel<<<3, 256, 0, stream>>>(W1, W2, W3, scales);
    pack_signs_kernel<<<(packN + 255) / 256, 256, 0, stream>>>(W1, W2, W3,
                                                               sw1, sw2, sw3);
    mlp_fused_kernel<<<rows / 16, 256, 0, stream>>>(
        x, sw1, sw2, sw3, b1, b2, b3, scales,
        s_in, s1_in, s1_out, s2_in, s2_out, s3_in, s3_out, out);
}
